// ConvBlock_4346506903658
// MI455X (gfx1250) — matrix operand order repaired and checked by reading, not compiled, not run on hardware
//
#include <hip/hip_runtime.h>

// ---------------- types (gfx1250, wave32) ----------------
typedef __bf16        v8bf  __attribute__((ext_vector_type(8)));
typedef __bf16        v16bf __attribute__((ext_vector_type(16)));
typedef float         v8f   __attribute__((ext_vector_type(8)));
typedef unsigned int  v4u   __attribute__((ext_vector_type(4)));
typedef int           v8i   __attribute__((ext_vector_type(8)));
typedef int           v4i   __attribute__((ext_vector_type(4)));

// Problem constants
#define BB   32
#define CC   128
#define HH   64
#define WW   64
#define OUTC 256
#define OH   62
#define OW   62
#define KTOT 1152      // 9 taps * 128 channels, reordered: k' = tap*128 + c
#define KT   32        // K tile (one bf16 WMMA depth)
#define APAD 40        // A-tile row pitch in bf16 elems (80B, 16B aligned, conflict-free)

// LDS layout (bytes) inside one 67584-B buffer
#define A_OFF      0            // 64*APAD*2 = 5120
#define B_OFF      5120         // 2 x 20480 double-buffered B tiles
#define BPITCH     80           // LDS pitch per oc row (32 bf16 + 16B pad)
#define BTILE_B    (OUTC * BPITCH)   // 20480
#define STATS_OFF  (B_OFF + 2 * BTILE_B)   // 46080, 1024 B
#define LDS_BYTES  (256 * 65 * 4 + 1024)   // 67584 (stage region reuses everything)

#if defined(__has_builtin)
#if __has_builtin(__builtin_amdgcn_tensor_load_to_lds)
#define HAVE_TDM 1
#endif
#endif

__device__ __forceinline__ void wait_tensorcnt0() {
#if defined(__has_builtin) && __has_builtin(__builtin_amdgcn_s_wait_tensorcnt)
    __builtin_amdgcn_s_wait_tensorcnt((short)0);
#else
    asm volatile("s_wait_tensorcnt 0x0" ::: "memory");
#endif
}

// round-to-nearest-even f32 -> bf16 bits
__device__ __forceinline__ unsigned short f2bf_bits(float f) {
    unsigned u = __builtin_bit_cast(unsigned, f);
    u += 0x7FFFu + ((u >> 16) & 1u);
    return (unsigned short)(u >> 16);
}

#ifdef HAVE_TDM
// Issue one TDM load: 2D tile (32 k-elems x 256 oc rows) of bf16 weights into LDS,
// TDM inserts 16B padding after each 64B row -> 80B LDS pitch.
__device__ __forceinline__ void tdm_load_btile(const unsigned short* Wt, int k0,
                                               unsigned ldsByteAddr) {
    unsigned long long ga = (unsigned long long)(const void*)Wt
                          + (unsigned long long)k0 * 2ull;
    v4u g0;
    g0[0] = 1u;                                              // count=1, no gather
    g0[1] = ldsByteAddr;                                     // lds_addr (bytes)
    g0[2] = (unsigned)(ga & 0xFFFFFFFFull);                  // global addr lo
    g0[3] = (unsigned)((ga >> 32) & 0x01FFFFFFull) | (2u << 30); // addr[56:32] | type=2
    v8i g1;
    g1[0] = (1 << 16)       // data_size = 2 bytes
          | (1 << 20)       // pad_enable
          | (3 << 22)       // pad_interval code 3 = every 16 DWORDs (one 64B row)
          | (3 << 25);      // pad_amount   code 3 = 4 DWORDs (16B) -> 80B pitch
    g1[1] = (KTOT & 0xFFFF) << 16;   // tensor_dim0[15:0] @ bits 63:48
    g1[2] = (OUTC << 16);            // tensor_dim0 hi=0 | tensor_dim1[15:0] @ 95:80
    g1[3] = (KT << 16);              // tensor_dim1 hi=0 | tile_dim0=32 @ 127:112
    g1[4] = OUTC;                    // tile_dim1=256 @ 143:128, tile_dim2=0
    g1[5] = KTOT;                    // tensor_dim0_stride lo32 (elements)
    g1[6] = 0;                       // stride hi / tensor_dim1_stride (2D: unused)
    g1[7] = 0;
    v4i z4 = {0, 0, 0, 0};
#if defined(__clang_major__) && (__clang_major__ >= 23)
    // clang-23 / therock: 6-arg form (extra int32x8 operand, zero-filled)
    v8i z8 = {0, 0, 0, 0, 0, 0, 0, 0};
    __builtin_amdgcn_tensor_load_to_lds(g0, g1, z4, z4, z8, 0);
#else
    // ROCm 7.2 / clang-22: 5-arg form
    __builtin_amdgcn_tensor_load_to_lds(g0, g1, z4, z4, 0);
#endif
}
#endif

// ---------------- kernel 1: per-channel BN stats -> (scale, bias) ----------------
__global__ void bn_stats_kernel(const float* __restrict__ x,
                                const float* __restrict__ gamma,
                                const float* __restrict__ beta,
                                float2* __restrict__ stats) {
    __shared__ float rs[256];
    __shared__ float rss[256];
    const int c   = blockIdx.x;
    const int tid = threadIdx.x;
    float s = 0.f, ss = 0.f;
    const float* xp = x + (size_t)c * (HH * WW);
    for (int idx = tid; idx < BB * HH * WW; idx += 256) {
        int n = idx >> 12;
        int i = idx & 4095;
        float v = xp[(size_t)n * (CC * HH * WW) + i];
        s  += v;
        ss += v * v;
    }
    rs[tid] = s; rss[tid] = ss;
    __syncthreads();
    for (int off = 128; off > 0; off >>= 1) {
        if (tid < off) { rs[tid] += rs[tid + off]; rss[tid] += rss[tid + off]; }
        __syncthreads();
    }
    if (tid == 0) {
        const float invN = 1.f / (float)(BB * HH * WW);
        float mean = rs[0] * invN;
        float var  = rss[0] * invN - mean * mean;
        float rsq  = rsqrtf(var + 1e-5f);
        float sc   = gamma[c] * rsq;
        stats[c]   = make_float2(sc, beta[c] - mean * sc);
    }
}

// ---- kernel 2: weight (korig, oc) f32 -> Wt[oc][k'] bf16, k' = tap*128 + c ----
__global__ void wcvt_kernel(const float* __restrict__ w,
                            unsigned short* __restrict__ Wt) {
    const int kp = blockIdx.x;            // 0..1151 (reordered index)
    const int oc = threadIdx.x;           // 0..255
    const int t  = kp >> 7;               // tap 0..8
    const int c  = kp & 127;
    const int ko = c * 9 + t;             // original row of weight
    Wt[(size_t)oc * KTOT + kp] = f2bf_bits(w[(size_t)ko * OUTC + oc]);
}

// ---------------- kernel 3: fused BN+ReLU + implicit-GEMM conv + residual ----------------
// grid (62, 32): blockIdx.x = oh, blockIdx.y = n. 256 threads = 8 waves.
__global__ __launch_bounds__(256)
void conv_wmma_kernel(const float* __restrict__ x,
                      const float2* __restrict__ stats,
                      const unsigned short* __restrict__ Wt,
                      float* __restrict__ out) {
    __shared__ __align__(16) unsigned char lds_buf[LDS_BYTES];
    unsigned short* Atile   = (unsigned short*)(lds_buf + A_OFF);
    float2*         stCache = (float2*)(lds_buf + STATS_OFF);
    float*          stage   = (float*)lds_buf;   // epilogue reuses whole buffer

    const int tid  = threadIdx.x;
    const int lane = tid & 31;
    const int wave = tid >> 5;
    const int oh   = blockIdx.x;
    const int n    = blockIdx.y;

    const int mb     = wave & 3;           // M block (ow/16)
    const int ocBase = (wave >> 2) * 128;  // 0 or 128

    if (tid < CC) stCache[tid] = stats[tid];

    // A-tile build assignment: thread -> row m (0..63), kk group of 8
    const int am = tid & 63;
    const int ag = tid >> 6;               // 0..3

    v8f acc[8];
    #pragma unroll
    for (int i = 0; i < 8; ++i) acc[i] = (v8f)0.f;

    const int arow = lane & 15;
    const int kb8  = (lane >> 4) * 16;     // A-fragment K base (contiguous 16 per half)
    const int kb16 = (lane >> 4) * 16;     // B-fragment K base

    const int nTiles = KTOT / KT;          // 36
#ifdef HAVE_TDM
    const unsigned ldsBase = (unsigned)(size_t)(void*)lds_buf;
    if (wave == 0) tdm_load_btile(Wt, 0, ldsBase + B_OFF);
#endif

    for (int it = 0; it < nTiles; ++it) {
        const int k0 = it * KT;
        const int t  = k0 >> 7;            // tap, uniform over the whole K tile
        const int c0 = k0 & 127;
        const int r  = t / 3;
        const int s  = t - r * 3;
        const unsigned char* bbase = lds_buf + B_OFF + (unsigned)(it & 1) * BTILE_B;

#ifdef HAVE_TDM
        if (wave == 0) wait_tensorcnt0();          // current B tile landed
        __syncthreads();                           // everyone sees it; A tile free
        if (wave == 0 && it + 1 < nTiles)
            tdm_load_btile(Wt, k0 + KT,
                           ldsBase + B_OFF + (unsigned)((it + 1) & 1) * BTILE_B);
#else
        __syncthreads();
        {   // cooperative B-tile copy: thread tid owns oc row tid (64B -> 80B pitch)
            const uint4* src = (const uint4*)(Wt + (size_t)tid * KTOT + k0);
            uint4* dst = (uint4*)(lds_buf + B_OFF + (unsigned)(it & 1) * BTILE_B
                                  + (unsigned)tid * BPITCH);
            #pragma unroll
            for (int i = 0; i < 4; ++i) dst[i] = src[i];
        }
#endif

        // ---- im2col + BN + ReLU + bf16: 8 elements/thread into Atile[m][kk] ----
        {
            int iw = am + s; if (iw > WW - 1) iw = WW - 1;       // masked rows clamp
            const float* xb = x + (((size_t)(n * CC + c0)) << 12) + ((oh + r) << 6) + iw;
            const float* xk = xb + ((size_t)(ag * 8) << 12);
            #pragma unroll
            for (int j = 0; j < 8; j += 2) {
                const int kk = ag * 8 + j;
                const float2 sb0 = stCache[c0 + kk];
                const float2 sb1 = stCache[c0 + kk + 1];
                const float v0 = xk[(size_t)j << 12];
                const float v1 = xk[(size_t)(j + 1) << 12];
                const float h0 = fmaxf(fmaf(v0, sb0.x, sb0.y), 0.f);
                const float h1 = fmaxf(fmaf(v1, sb1.x, sb1.y), 0.f);
                unsigned pk = (unsigned)f2bf_bits(h0)
                            | ((unsigned)f2bf_bits(h1) << 16);
                *(unsigned*)&Atile[am * APAD + kk] = pk;
            }
            // prefetch next K tile's activation line (global_prefetch_b8)
            if (it + 1 < nTiles) {
                const int k0n = k0 + KT;
                const int tn  = k0n >> 7;
                const int c0n = k0n & 127;
                const int rn  = tn / 3;
                const int sn  = tn - rn * 3;
                int iwn = am + sn; if (iwn > WW - 1) iwn = WW - 1;
                __builtin_prefetch(x + (((size_t)(n * CC + c0n + ag * 8)) << 12)
                                     + ((oh + rn) << 6) + iwn, 0, 3);
            }
        }
        __syncthreads();

        // ---- A fragment (16x32 bf16, this code's contiguous k order: lane%16 = M, lane/16 = K half) ----
        const unsigned short* ap = &Atile[(mb * 16 + arow) * APAD];
        v8bf alo = *(const v8bf*)(const void*)(ap + kb8);
        v8bf ahi = *(const v8bf*)(const void*)(ap + 8 + kb8);
        v16bf a;
        #pragma unroll
        for (int i = 0; i < 8; ++i) { a[i] = alo[i]; a[i + 8] = ahi[i]; }

        // ---- 8 B fragments (32x16 bf16) from the TDM-staged LDS tile + WMMA ----
        #pragma unroll
        for (int nb = 0; nb < 8; ++nb) {
            const int oc = ocBase + nb * 16 + (lane & 15);
            const unsigned short* wp =
                (const unsigned short*)(bbase + (unsigned)oc * BPITCH) + kb16;
            v8bf blo = *(const v8bf*)(const void*)wp;
            v8bf bhi = *(const v8bf*)(const void*)(wp + 8);
            v16bf b;
            #pragma unroll
            for (int i = 0; i < 8; ++i) { b[i] = blo[i]; b[i + 8] = bhi[i]; }
            acc[nb] = __builtin_amdgcn_wmma_f32_16x16x32_bf16(
                false, a, false, b, (short)0, acc[nb], false, false);
        }
    }

    // ---- epilogue: D tiles -> LDS transpose, then coalesced residual+store ----
    __syncthreads();
    {
        const int mrowbase = mb * 16 + ((lane >> 4) * 8);
        #pragma unroll
        for (int nb = 0; nb < 8; ++nb) {
            const int oc = ocBase + nb * 16 + (lane & 15);
            #pragma unroll
            for (int v = 0; v < 8; ++v)
                stage[oc * 65 + mrowbase + v] = acc[nb][v];
        }
    }
    __syncthreads();

    for (int i = 0; i < 64; ++i) {
        const int idx = i * 256 + tid;          // 256 oc x 64 m
        const int oc  = idx >> 6;
        const int m   = idx & 63;
        if (m < OW) {
            float g = stage[oc * 65 + m];
            if (oc < CC)
                g += x[(((size_t)(n * CC + oc)) << 12) + ((oh + 1) << 6) + (m + 1)];
            out[(((size_t)(n * OUTC + oc)) * OH + oh) * OW + m] = g;
        }
    }
}

extern "C" void kernel_launch(void* const* d_in, const int* in_sizes, int n_in,
                              void* d_out, int out_size, void* d_ws, size_t ws_size,
                              hipStream_t stream) {
    const float* x      = (const float*)d_in[0];
    const float* gamma  = (const float*)d_in[1];
    const float* beta   = (const float*)d_in[2];
    const float* weight = (const float*)d_in[3];
    float* out          = (float*)d_out;

    float2* stats      = (float2*)d_ws;                          // 1 KB
    unsigned short* Wt = (unsigned short*)((char*)d_ws + 1024);  // 576 KB

    bn_stats_kernel<<<CC, 256, 0, stream>>>(x, gamma, beta, stats);
    wcvt_kernel<<<KTOT, 256, 0, stream>>>(weight, Wt);
    conv_wmma_kernel<<<dim3(OH, BB), 256, 0, stream>>>(x, stats, Wt, out);
}
